// WindowAttention_67199058313372
// MI455X (gfx1250) — matrix operand order repaired and checked by reading, not compiled, not run on hardware
//
#include <hip/hip_runtime.h>

typedef __attribute__((ext_vector_type(16))) __bf16 bf16x16;
typedef __attribute__((ext_vector_type(8)))  __bf16 bf16x8;
typedef __attribute__((ext_vector_type(8)))  float  f32x8;

union AFrag { bf16x16 v; bf16x8 h[2]; };

// ---------------------------------------------------------------------------
// f32 -> bf16 conversion (grid-stride)
// ---------------------------------------------------------------------------
__global__ void cvt_bf16(const float* __restrict__ in, __bf16* __restrict__ out, long n) {
    long i = (long)blockIdx.x * blockDim.x + threadIdx.x;
    long stride = (long)gridDim.x * blockDim.x;
    for (; i < n; i += stride) out[i] = (__bf16)in[i];
}

// ---------------------------------------------------------------------------
// QKV GEMM: [M,256] x [256,768]^T + bias -> q[b,h,t,d] (scaled), k[b,h,t,d],
// vT[b,h,d,t]  (all bf16).  Block = 128x64 tile, 8 waves of 32x32.
// ---------------------------------------------------------------------------
__global__ __launch_bounds__(256)
void gemm_qkv(const __bf16* __restrict__ xb, const __bf16* __restrict__ w,
              const float* __restrict__ bias,
              __bf16* __restrict__ q, __bf16* __restrict__ kk, __bf16* __restrict__ vt)
{
    const int lane = threadIdx.x & 31;
    const int wave = threadIdx.x >> 5;
    const int wm = wave >> 1, wn = wave & 1;
    const int m0 = blockIdx.x * 128 + wm * 32;
    const int n0 = blockIdx.y * 64 + wn * 32;
    const int lrow = lane & 15, half = lane >> 4;

    f32x8 acc[2][2] = {};

    for (int k0 = 0; k0 < 256; k0 += 32) {
        AFrag a[2];
#pragma unroll
        for (int i = 0; i < 2; ++i) {
            const __bf16* pa = xb + (size_t)(m0 + 16 * i + lrow) * 256 + k0 + half * 16;
            a[i].h[0] = *(const bf16x8*)pa;
            a[i].h[1] = *(const bf16x8*)(pa + 8);
        }
        bf16x16 bfr[2];
#pragma unroll
        for (int j = 0; j < 2; ++j) {
            const __bf16* pb = w + (size_t)(n0 + 16 * j + lrow) * 256 + k0 + half * 16;
            bfr[j] = *(const bf16x16*)pb;
        }
#pragma unroll
        for (int i = 0; i < 2; ++i)
#pragma unroll
            for (int j = 0; j < 2; ++j)
                acc[i][j] = __builtin_amdgcn_wmma_f32_16x16x32_bf16(
                    false, a[i].v, false, bfr[j], (short)0, acc[i][j], false, false);
    }

    const float qscale = 0.17677669529663689f; // 32^-0.5
#pragma unroll
    for (int j = 0; j < 2; ++j) {
        const int col = n0 + 16 * j + lrow;
        const float bv = bias[col];
#pragma unroll
        for (int i = 0; i < 2; ++i) {
            const int rbase = m0 + 16 * i + 8 * half;      // multiple of 8
            const int bwin = rbase >> 6;                    // constant over r
            const int tok0 = rbase & 63;
            if (col < 256) {                                // Q (pre-scaled)
                const int h = col >> 5, d = col & 31;
                __bf16* dst = q + (((size_t)bwin * 8 + h) * 64) * 32 + d;
#pragma unroll
                for (int r = 0; r < 8; ++r)
                    dst[(size_t)(tok0 + r) * 32] = (__bf16)((acc[i][j][r] + bv) * qscale);
            } else if (col < 512) {                         // K
                const int f = col - 256, h = f >> 5, d = f & 31;
                __bf16* dst = kk + (((size_t)bwin * 8 + h) * 64) * 32 + d;
#pragma unroll
                for (int r = 0; r < 8; ++r)
                    dst[(size_t)(tok0 + r) * 32] = (__bf16)(acc[i][j][r] + bv);
            } else {                                        // V, stored transposed
                const int f = col - 512, h = f >> 5, d = f & 31;
                bf16x8 pk;
#pragma unroll
                for (int r = 0; r < 8; ++r) pk[r] = (__bf16)(acc[i][j][r] + bv);
                *(bf16x8*)(vt + ((((size_t)bwin * 8 + h) * 32 + d) * 64 + tok0)) = pk;
            }
        }
    }
}

// ---------------------------------------------------------------------------
// Fused attention: one block per (window, head); 4 waves; wave w owns score
// rows 16w..16w+15.  QK^T (4 WMMAs) + rpb + mask + softmax + PV (4 WMMAs).
// P goes through LDS to convert C/D layout -> A layout in bf16.
// ---------------------------------------------------------------------------
__global__ __launch_bounds__(128)
void attn_kernel(const __bf16* __restrict__ q, const __bf16* __restrict__ kk,
                 const __bf16* __restrict__ vt, const float* __restrict__ rpb,
                 const float* __restrict__ maskp, __bf16* __restrict__ ao, int nW)
{
    __shared__ __bf16 smem[4][16][64];   // 8 KB: per-wave 16x64 bf16 P tile
    const int b = blockIdx.x, h = blockIdx.y;
    const int lane = threadIdx.x & 31;
    const int w = threadIdx.x >> 5;
    const int lrow = lane & 15, half = lane >> 4;

    const size_t bh = (size_t)b * 8 + h;
    const __bf16* qh = q + bh * 64 * 32;
    const __bf16* kh = kk + bh * 64 * 32;
    const __bf16* vh = vt + bh * 32 * 64;

    // Q A-fragment: rows 16w..16w+15, K = full hd (32)
    AFrag aq;
    {
        const __bf16* pa = qh + (size_t)(16 * w + lrow) * 32 + half * 16;
        aq.h[0] = *(const bf16x8*)pa;
        aq.h[1] = *(const bf16x8*)(pa + 8);
    }

    f32x8 acc[4];
#pragma unroll
    for (int f = 0; f < 4; ++f) {
        const __bf16* pb = kh + (size_t)(f * 16 + lrow) * 32 + half * 16;
        bf16x16 bk = *(const bf16x16*)pb;
        f32x8 z = {};
        acc[f] = __builtin_amdgcn_wmma_f32_16x16x32_bf16(
            false, aq.v, false, bk, (short)0, z, false, false);
    }

    // relative position bias + shift-window mask
    const float* mk = maskp + (size_t)(b % nW) * 4096;
#pragma unroll
    for (int f = 0; f < 4; ++f) {
        const int n = f * 16 + lrow;
        const int ni = n >> 3, nj = n & 7;
#pragma unroll
        for (int r = 0; r < 8; ++r) {
            const int m = 16 * w + r + 8 * half;
            const int mi = m >> 3, mj = m & 7;
            const int idx = (mi - ni + 7) * 15 + (mj - nj + 7);
            acc[f][r] += rpb[idx * 8 + h] + mk[m * 64 + n];
        }
    }

    // softmax: rows live in 16-lane halves -> xor reductions 1,2,4,8
#pragma unroll
    for (int r = 0; r < 8; ++r) {
        float mx = fmaxf(fmaxf(acc[0][r], acc[1][r]), fmaxf(acc[2][r], acc[3][r]));
#pragma unroll
        for (int s = 1; s <= 8; s <<= 1) mx = fmaxf(mx, __shfl_xor(mx, s, 32));
        float sum = 0.f;
#pragma unroll
        for (int f = 0; f < 4; ++f) {
            float e = __expf(acc[f][r] - mx);
            acc[f][r] = e;
            sum += e;
        }
#pragma unroll
        for (int s = 1; s <= 8; s <<= 1) sum += __shfl_xor(sum, s, 32);
        const float inv = 1.f / sum;
#pragma unroll
        for (int f = 0; f < 4; ++f) acc[f][r] *= inv;
    }

    // C/D layout -> row-major bf16 in LDS
#pragma unroll
    for (int f = 0; f < 4; ++f)
#pragma unroll
        for (int r = 0; r < 8; ++r)
            smem[w][r + 8 * half][f * 16 + lrow] = (__bf16)acc[f][r];
    __syncthreads();

    // P @ V: K=64 in two steps of 32; vT gives contiguous B-fragments
    f32x8 o[2] = {};
#pragma unroll
    for (int kb = 0; kb < 2; ++kb) {
        AFrag ap;
        const __bf16* pa = &smem[w][lrow][kb * 32 + half * 16];
        ap.h[0] = *(const bf16x8*)pa;
        ap.h[1] = *(const bf16x8*)(pa + 8);
#pragma unroll
        for (int nb = 0; nb < 2; ++nb) {
            const __bf16* pb = vh + (size_t)(nb * 16 + lrow) * 64 + kb * 32 + half * 16;
            bf16x16 bvf = *(const bf16x16*)pb;
            o[nb] = __builtin_amdgcn_wmma_f32_16x16x32_bf16(
                false, ap.v, false, bvf, (short)0, o[nb], false, false);
        }
    }

    // store to [B_*64, 256] bf16; channel = h*32 + d
#pragma unroll
    for (int nb = 0; nb < 2; ++nb) {
        const int d = nb * 16 + lrow;
        const int c = h * 32 + d;
#pragma unroll
        for (int r = 0; r < 8; ++r) {
            const int t = 16 * w + r + 8 * half;
            ao[((size_t)b * 64 + t) * 256 + c] = (__bf16)o[nb][r];
        }
    }
}

// ---------------------------------------------------------------------------
// Output projection: [M,256] x [256,256]^T + bias -> f32 out
// ---------------------------------------------------------------------------
__global__ __launch_bounds__(256)
void gemm_proj(const __bf16* __restrict__ ab, const __bf16* __restrict__ w,
               const float* __restrict__ bias, float* __restrict__ out)
{
    const int lane = threadIdx.x & 31;
    const int wave = threadIdx.x >> 5;
    const int wm = wave >> 1, wn = wave & 1;
    const int m0 = blockIdx.x * 128 + wm * 32;
    const int n0 = blockIdx.y * 64 + wn * 32;
    const int lrow = lane & 15, half = lane >> 4;

    f32x8 acc[2][2] = {};

    for (int k0 = 0; k0 < 256; k0 += 32) {
        AFrag a[2];
#pragma unroll
        for (int i = 0; i < 2; ++i) {
            const __bf16* pa = ab + (size_t)(m0 + 16 * i + lrow) * 256 + k0 + half * 16;
            a[i].h[0] = *(const bf16x8*)pa;
            a[i].h[1] = *(const bf16x8*)(pa + 8);
        }
        bf16x16 bfr[2];
#pragma unroll
        for (int j = 0; j < 2; ++j) {
            const __bf16* pb = w + (size_t)(n0 + 16 * j + lrow) * 256 + k0 + half * 16;
            bfr[j] = *(const bf16x16*)pb;
        }
#pragma unroll
        for (int i = 0; i < 2; ++i)
#pragma unroll
            for (int j = 0; j < 2; ++j)
                acc[i][j] = __builtin_amdgcn_wmma_f32_16x16x32_bf16(
                    false, a[i].v, false, bfr[j], (short)0, acc[i][j], false, false);
    }

#pragma unroll
    for (int j = 0; j < 2; ++j) {
        const int col = n0 + 16 * j + lrow;
        const float bv = bias[col];
#pragma unroll
        for (int i = 0; i < 2; ++i) {
            const int rbase = m0 + 16 * i + 8 * half;
#pragma unroll
            for (int r = 0; r < 8; ++r)
                out[(size_t)(rbase + r) * 256 + col] = acc[i][j][r] + bv;
        }
    }
}

// ---------------------------------------------------------------------------
extern "C" void kernel_launch(void* const* d_in, const int* in_sizes, int n_in,
                              void* d_out, int out_size, void* d_ws, size_t ws_size,
                              hipStream_t stream) {
    const float* x      = (const float*)d_in[0];
    const float* mask   = (const float*)d_in[1];
    const float* qkv_w  = (const float*)d_in[2];
    const float* qkv_b  = (const float*)d_in[3];
    const float* proj_w = (const float*)d_in[4];
    const float* proj_b = (const float*)d_in[5];
    const float* rpb    = (const float*)d_in[6];

    const long Bwin = (long)in_sizes[0] / (64 * 256);   // 2048
    const int  nW   = in_sizes[1] / 4096;               // 1024
    const long M    = Bwin * 64;                        // 131072

    char* wsp = (char*)d_ws;
    size_t off = 0;
    auto alloc = [&](size_t bytes) -> void* {
        void* p = wsp + off;
        off += (bytes + 255) & ~(size_t)255;
        return p;
    };
    __bf16* xb = (__bf16*)alloc((size_t)M * 256 * 2);
    __bf16* wq = (__bf16*)alloc((size_t)768 * 256 * 2);
    __bf16* wp = (__bf16*)alloc((size_t)256 * 256 * 2);
    __bf16* qb = (__bf16*)alloc((size_t)M * 256 * 2);
    __bf16* kb = (__bf16*)alloc((size_t)M * 256 * 2);
    __bf16* vb = (__bf16*)alloc((size_t)M * 256 * 2);
    __bf16* ab = (__bf16*)alloc((size_t)M * 256 * 2);
    (void)ws_size; (void)n_in; (void)out_size;

    cvt_bf16<<<dim3(4096), 256, 0, stream>>>(x, xb, M * 256);
    cvt_bf16<<<dim3(768), 256, 0, stream>>>(qkv_w, wq, 768L * 256);
    cvt_bf16<<<dim3(256), 256, 0, stream>>>(proj_w, wp, 256L * 256);

    gemm_qkv<<<dim3((unsigned)(M / 128), 12), 256, 0, stream>>>(xb, wq, qkv_b, qb, kb, vb);
    attn_kernel<<<dim3((unsigned)Bwin, 8), 128, 0, stream>>>(qb, kb, vb, rpb, mask, ab, nW);
    gemm_proj<<<dim3((unsigned)(M / 128), 4), 256, 0, stream>>>(ab, wp, proj_b, (float*)d_out);
}
